// STLVQE_21036749816335
// MI455X (gfx1250) — matrix operand order repaired and checked by reading, not compiled, not run on hardware
//
#include <hip/hip_runtime.h>

typedef __attribute__((ext_vector_type(16))) _Float16 v16h;
typedef __attribute__((ext_vector_type(8)))  _Float16 v8h;
typedef __attribute__((ext_vector_type(8)))  float    v8f;
typedef __attribute__((ext_vector_type(4)))  float    v4f;

static constexpr int HPIX = 192, WPIX = 192;
static constexpr size_t HW = (size_t)HPIX * WPIX;          // 36864
static constexpr size_t CUR_STRIDE = (size_t)4 * 32 * HW;  // per-rotation block
static constexpr size_t XALL_OFF   = 4 * CUR_STRIDE;
static constexpr size_t SUR_OFF    = XALL_OFF + 4 * CUR_STRIDE;
static constexpr size_t WGT_OFF    = SUR_OFF + (size_t)8 * 32 * HW;

__device__ __forceinline__ float lrelu_f(float v) { return v >= 0.f ? v : 0.1f * v; }
// forward value of lqs(): s_scale == s, y = round(clip(x/s, 0, hi)) * s
__device__ __forceinline__ float lqs_q(float v, float s, float hi) {
    float y = v / s;
    y = fminf(fmaxf(y, 0.f), hi);
    return rintf(y) * s;
}

// ---------------------------------------------------------------------------
// f32 -> f16 conversion (activations into workspace), 4 elems/thread
// ---------------------------------------------------------------------------
__global__ void cvt_f16_kernel(const float* __restrict__ src, _Float16* __restrict__ dst, int n4) {
    int i = blockIdx.x * blockDim.x + threadIdx.x;
    if (i < n4) {
        v4f v = *(const v4f*)(src + 4 * (size_t)i);
        _Float16 h0 = (_Float16)v.x, h1 = (_Float16)v.y, h2 = (_Float16)v.z, h3 = (_Float16)v.w;
        v8h* d = nullptr; (void)d;
        dst[4 * (size_t)i + 0] = h0;
        dst[4 * (size_t)i + 1] = h1;
        dst[4 * (size_t)i + 2] = h2;
        dst[4 * (size_t)i + 3] = h3;
    }
}

// ---------------------------------------------------------------------------
// Implicit-GEMM 3x3 conv (pad 1) with V_WMMA_F32_16X16X32_F16.
// C[Cout x pix] = W[Cout x (Cin*9)] * im2col[(Cin*9) x pix]
// Block: 256 thr = 8 waves (2 x 4); block tile = 32 Cout x 64 pixels.
// K-chunk = 64 -> 2 WMMAs per barrier pair.
// Staging is branch-free: clamped-address unconditional loads + select,
// batched into register arrays so loads pipeline (single wait per stage).
// MODE 0: bias+lrelu -> f16 NCHW                         (COUT=128)
// MODE 1: bias+lrelu+pixelshuffle(2) -> f16 [B][COUT/4][2H][2W] (COUT=512)
// MODE 2: bias only -> f32 feat, guard cout<COUT         (COUT=38)
// ---------------------------------------------------------------------------
template <int CIN, int COUT, int HH, int WW, int MODE>
__global__ __launch_bounds__(256) void conv3x3_wmma(
        const _Float16* __restrict__ in, const float* __restrict__ wgt,
        const float* __restrict__ bias, _Float16* __restrict__ outh,
        float* __restrict__ outf) {
    constexpr int HWL = HH * WW;
    constexpr bool COUT_ALIGNED = (COUT % 32) == 0;
    const int coutBase = blockIdx.x * 32;
    const int pixBase  = blockIdx.y * 64;
    const int b        = blockIdx.z;
    const int tid  = threadIdx.x;
    const int lane = tid & 31;
    const int wv   = tid >> 5;
    const int wrow = wv >> 2;   // 0..1  (16 couts each)
    const int wcol = wv & 3;    // 0..3  (16 pixels each)

    __shared__ _Float16 As[32][64];   // [cout_local][k]   4 KB
    __shared__ _Float16 Bs[64][64];   // [pix_local][k]    8 KB

    // A-staging role: thread covers row ai, k = ak0..ak0+7 (stride-9 f32 loads)
    const int ai  = tid >> 3;
    const int ak0 = (tid & 7) * 8;
    const int aco = coutBase + ai;
    const bool aok = COUT_ALIGNED || (aco < COUT);
    const int acoc = aok ? aco : 0;   // clamped row for branch-free loads

    // B-staging role: thread covers k = bk, pixels pg*16 .. pg*16+15
    const int bk = tid & 63;
    const int pg = tid >> 6;

    v8f acc = {0.f, 0.f, 0.f, 0.f, 0.f, 0.f, 0.f, 0.f};

    for (int tap = 0; tap < 9; ++tap) {
        const int ky = tap / 3, kx = tap % 3;
        for (int cin0 = 0; cin0 < CIN; cin0 += 64) {
            // speculative prefetch of the next weight chunk (global_prefetch_b8)
            __builtin_prefetch(wgt + ((size_t)coutBase * CIN + cin0 + 64) * 9, 0, 1);

            // ---- stage A: 32 couts x 64 k (branch-free, batched loads) ----
            {
                const float* wp = wgt + (((size_t)acoc * CIN + cin0 + ak0) * 3 + ky) * 3 + kx;
                float wvreg[8];
#pragma unroll
                for (int u = 0; u < 8; ++u) wvreg[u] = wp[(size_t)u * 9];
#pragma unroll
                for (int u = 0; u < 8; ++u)
                    As[ai][ak0 + u] = aok ? (_Float16)wvreg[u] : (_Float16)0.f;
            }
            // ---- stage B: 64 pixels x 64 k (branch-free, batched loads) ----
            {
                const _Float16* src = in + ((size_t)b * CIN + cin0 + bk) * HWL;
                _Float16 tv[16];
                bool okm[16];
#pragma unroll
                for (int u = 0; u < 16; ++u) {
                    int P = pixBase + pg * 16 + u;
                    int y = P / WW, x = P % WW;
                    int iy = y + ky - 1, ix = x + kx - 1;
                    bool ok = (iy >= 0) & (iy < HH) & (ix >= 0) & (ix < WW);
                    int off = ok ? iy * WW + ix : 0;
                    tv[u]  = src[off];
                    okm[u] = ok;
                }
#pragma unroll
                for (int u = 0; u < 16; ++u)
                    Bs[pg * 16 + u][bk] = okm[u] ? tv[u] : (_Float16)0.f;
            }
            __syncthreads();
            // ---- 2 x WMMA per stage (CDNA5 16-bit 16x32 A / 32x16 B layouts) ----
            const int hs = lane >> 4;
            const int ml = lane & 15;
#pragma unroll
            for (int kk = 0; kk < 64; kk += 32) {
                union { v16h v; v8h h[2]; } afr, bfr;
                afr.h[0] = *(const v8h*)&As[wrow * 16 + ml][kk + 8 * hs];
                afr.h[1] = *(const v8h*)&As[wrow * 16 + ml][kk + 16 + 8 * hs];
                bfr.h[0] = *(const v8h*)&Bs[wcol * 16 + ml][kk + 8 * hs];
                bfr.h[1] = *(const v8h*)&Bs[wcol * 16 + ml][kk + 16 + 8 * hs];
                acc = __builtin_amdgcn_wmma_f32_16x16x32_f16(
                        false, afr.v, false, bfr.v, (short)0, acc, false, false);
            }
            __syncthreads();
        }
    }

    // ---- epilogue: C layout, VGPR j: M=j (lanes 0-15), M=j+8 (lanes 16-31) ----
    const int hs = lane >> 4;
    const int n  = lane & 15;
    const int P  = pixBase + wcol * 16 + n;
    const int y  = P / WW, x = P % WW;
#pragma unroll
    for (int j = 0; j < 8; ++j) {
        int cout = coutBase + wrow * 16 + 8 * hs + j;
        float v = acc[j];
        if constexpr (MODE == 0) {
            v = lrelu_f(v + bias[cout]);
            outh[((size_t)b * COUT + cout) * HWL + P] = (_Float16)v;
        } else if constexpr (MODE == 1) {
            v = lrelu_f(v + bias[cout]);
            int co = cout >> 2, r1 = (cout >> 1) & 1, r2 = cout & 1;
            outh[(((size_t)b * (COUT / 4) + co) * (2 * HH) + (2 * y + r1)) * (2 * WW) + (2 * x + r2)] =
                (_Float16)v;
        } else {
            if (cout < COUT)
                outf[((size_t)b * COUT + cout) * HWL + P] = v + bias[cout];
        }
    }
}

// ---------------------------------------------------------------------------
// deform_sample: offsets = 10*tanh(feat[:, :36]) (batch-split), clamped bilinear
// samples[bb][tap][h][w], bb in 0..7, tap = ky*3+kx
// ---------------------------------------------------------------------------
__global__ void deform_kernel(const float* __restrict__ x, const float* __restrict__ feat,
                              float* __restrict__ samples) {
    int idx = blockIdx.x * blockDim.x + threadIdx.x;
    if (idx >= 8 * 9 * (int)HW) return;
    int p   = idx % (int)HW;
    int tap = (idx / (int)HW) % 9;
    int bb  = idx / (9 * (int)HW);
    int h = p / WPIX, w = p % WPIX;
    int fb = (bb < 4) ? bb : bb - 4;
    int cb = (bb < 4) ? 0 : 18;
    float offx = 10.f * tanhf(feat[((size_t)fb * 38 + cb + tap) * HW + p]);
    float offy = 10.f * tanhf(feat[((size_t)fb * 38 + cb + 9 + tap) * HW + p]);
    float px = (float)(h + 1) + (float)(tap / 3 - 1) + offx;
    float py = (float)(w + 1) + (float)(tap % 3 - 1) + offy;
    const float LIM = 193.f;   // Hp-1 = Wp-1 with pad=1
    float fx = floorf(px), fy = floorf(py);
    float qltx = fminf(fmaxf(fx, 0.f), LIM),       qlty = fminf(fmaxf(fy, 0.f), LIM);
    float qrbx = fminf(fmaxf(fx + 1.f, 0.f), LIM), qrby = fminf(fmaxf(fy + 1.f, 0.f), LIM);
    float pcx  = fminf(fmaxf(px, 0.f), LIM),       pcy  = fminf(fmaxf(py, 0.f), LIM);
    const float* xb = x + (size_t)bb * HW;
    auto xp = [&](float qx, float qy) -> float {
        int i = (int)qx - 1, j = (int)qy - 1;
        bool ok = (i >= 0) & (i < HPIX) & (j >= 0) & (j < WPIX);
        float v = xb[ok ? i * WPIX + j : 0];
        return ok ? v : 0.f;
    };
    float glt = (1.f + (qltx - pcx)) * (1.f + (qlty - pcy));
    float grb = (1.f - (qrbx - pcx)) * (1.f - (qrby - pcy));
    float glb = (1.f + (qltx - pcx)) * (1.f - (qrby - pcy));
    float grt = (1.f - (qrbx - pcx)) * (1.f + (qlty - pcy));
    samples[idx] = glt * xp(qltx, qlty) + grb * xp(qrbx, qrby) +
                   glb * xp(qltx, qrby) + grt * xp(qrbx, qlty);
}

// softmax over feat channels 36,37 -> weights [4][2][H][W]
__global__ void weights_kernel(const float* __restrict__ feat, float* __restrict__ out) {
    int idx = blockIdx.x * blockDim.x + threadIdx.x;
    if (idx >= 4 * (int)HW) return;
    int p = idx % (int)HW, b = idx / (int)HW;
    float f0 = feat[((size_t)b * 38 + 36) * HW + p];
    float f1 = feat[((size_t)b * 38 + 37) * HW + p];
    float m = fmaxf(f0, f1);
    float e0 = expf(f0 - m), e1 = expf(f1 - m), s = e0 + e1;
    out[WGT_OFF + ((size_t)b * 2 + 0) * HW + p] = e0 / s;
    out[WGT_OFF + ((size_t)b * 2 + 1) * HW + p] = e1 / s;
}

// x_cur_out[r]: rot90(x_current,r) edge-padded, lqs(s_df,0,127), 2x2 conv cc, lrelu
__global__ void cur_kernel(const float* __restrict__ xc, const float* __restrict__ ccw,
                           const float* __restrict__ ccb, const float* __restrict__ sdf,
                           float* __restrict__ out) {
    int idx = blockIdx.x * blockDim.x + threadIdx.x;
    if (idx >= 16 * (int)HW) return;
    int p = idx % (int)HW;
    int b = (idx / (int)HW) & 3;
    int r = idx / (4 * (int)HW);
    float s = sdf[0];
    int i = p / WPIX, j = p % WPIX;
    auto xr = [&](int ii, int jj) -> float {  // edge-pad then rotate-index
        ii = ii > HPIX - 1 ? HPIX - 1 : ii;
        jj = jj > WPIX - 1 ? WPIX - 1 : jj;
        int a, c;
        switch (r) {
            case 0:  a = ii;            c = jj;            break;
            case 1:  a = jj;            c = HPIX - 1 - ii; break;
            case 2:  a = HPIX - 1 - ii; c = WPIX - 1 - jj; break;
            default: a = WPIX - 1 - jj; c = ii;            break;
        }
        return xc[(size_t)b * HW + a * WPIX + c];
    };
    float v00 = lqs_q(xr(i, j), s, 127.f);
    float v01 = lqs_q(xr(i, j + 1), s, 127.f);
    float v10 = lqs_q(xr(i + 1, j), s, 127.f);
    float v11 = lqs_q(xr(i + 1, j + 1), s, 127.f);
    size_t base = (size_t)r * CUR_STRIDE + (size_t)b * 32 * HW + p;
    for (int o = 0; o < 32; ++o) {
        float v = ccw[o * 4 + 0] * v00 + ccw[o * 4 + 1] * v01 +
                  ccw[o * 4 + 2] * v10 + ccw[o * 4 + 3] * v11 + ccb[o];
        out[base + (size_t)o * HW] = lrelu_f(v);
    }
}

// x_all[r]: fold of rotated tap column, lqs(s_temp,0,20), (2,1)-stride conv ct, lrelu
__global__ void xall_kernel(const float* __restrict__ xc, const float* __restrict__ samples,
                            const float* __restrict__ ctw, const float* __restrict__ ctb,
                            const float* __restrict__ stp, float* __restrict__ out) {
    int idx = blockIdx.x * blockDim.x + threadIdx.x;
    if (idx >= 16 * (int)HW) return;
    int p = idx % (int)HW;
    int b = (idx / (int)HW) & 3;
    int r = idx / (4 * (int)HW);
    float s = stp[0];
    int h = p / WPIX, w = p % WPIX;
    // rotated tap (1+si, 1) mapped back to original (oy, ox)
    int oy[2], ox[2];
    switch (r) {
        case 0:  oy[0] = 1; ox[0] = 1; oy[1] = 2; ox[1] = 1; break;
        case 1:  oy[0] = 1; ox[0] = 1; oy[1] = 1; ox[1] = 0; break;
        case 2:  oy[0] = 1; ox[0] = 1; oy[1] = 0; ox[1] = 1; break;
        default: oy[0] = 1; ox[0] = 1; oy[1] = 1; ox[1] = 2; break;
    }
    float q[3][2];
    for (int si = 0; si < 2; ++si) {
        int ty = oy[si], tx = ox[si], tap = ty * 3 + tx;
        int ii = h + ty - 1, jj = w + tx - 1;
        bool ok = (ii >= 0) & (ii < HPIX) & (jj >= 0) & (jj < WPIX);
        float v0 = xc[(size_t)b * HW + (ok ? ii * WPIX + jj : 0)];
        v0 = ok ? v0 : 0.f;
        q[0][si] = lqs_q(v0, s, 20.f);
        q[1][si] = lqs_q(samples[((size_t)b * 9 + tap) * HW + p], s, 20.f);
        q[2][si] = lqs_q(samples[((size_t)(b + 4) * 9 + tap) * HW + p], s, 20.f);
    }
    size_t base = XALL_OFF + (size_t)r * CUR_STRIDE + (size_t)b * 32 * HW + p;
    for (int o = 0; o < 32; ++o) {
        float acc = ctb[o];
        for (int ic = 0; ic < 3; ++ic)
            for (int si = 0; si < 2; ++si)
                acc += ctw[(o * 3 + ic) * 2 + si] * q[ic][si];
        out[base + (size_t)o * HW] = lrelu_f(acc);
    }
}

// surround: strided(3)+dilated(2) 2x2 conv over lqs(x_offset) == 4 taps of samples
__global__ void surround_kernel(const float* __restrict__ samples, const float* __restrict__ csw,
                                const float* __restrict__ csb, const float* __restrict__ sdf,
                                float* __restrict__ out) {
    int idx = blockIdx.x * blockDim.x + threadIdx.x;
    if (idx >= 8 * (int)HW) return;
    int p = idx % (int)HW, b8 = idx / (int)HW;
    float s = sdf[0];
    float q[4];
    for (int di = 0; di < 2; ++di)
        for (int dj = 0; dj < 2; ++dj) {
            int tap = (2 * di) * 3 + 2 * dj;   // ky = 2*di, kx = 2*dj
            q[di * 2 + dj] = lqs_q(samples[((size_t)b8 * 9 + tap) * HW + p], s, 127.f);
        }
    size_t base = SUR_OFF + (size_t)b8 * 32 * HW + p;
    for (int o = 0; o < 32; ++o) {
        float v = csw[o * 4 + 0] * q[0] + csw[o * 4 + 1] * q[1] +
                  csw[o * 4 + 2] * q[2] + csw[o * 4 + 3] * q[3] + csb[o];
        out[base + (size_t)o * HW] = lrelu_f(v);
    }
}

// ---------------------------------------------------------------------------
extern "C" void kernel_launch(void* const* d_in, const int* in_sizes, int n_in,
                              void* d_out, int out_size, void* d_ws, size_t ws_size,
                              hipStream_t stream) {
    (void)in_sizes; (void)n_in; (void)out_size; (void)ws_size;
    const float* x    = (const float*)d_in[0];   // [8,1,192,192]
    const float* ef   = (const float*)d_in[1];   // [4,192,96,96]
    const float* xcur = (const float*)d_in[2];   // [4,1,192,192]
    const float* cf0w = (const float*)d_in[3];  const float* cf0b = (const float*)d_in[4];
    const float* cf1w = (const float*)d_in[5];  const float* cf1b = (const float*)d_in[6];
    const float* cf2w = (const float*)d_in[7];  const float* cf2b = (const float*)d_in[8];
    const float* psw  = (const float*)d_in[9];  const float* psb  = (const float*)d_in[10];
    const float* cf3w = (const float*)d_in[11]; const float* cf3b = (const float*)d_in[12];
    const float* ccw  = (const float*)d_in[13]; const float* ccb  = (const float*)d_in[14];
    const float* ctw  = (const float*)d_in[15]; const float* ctb  = (const float*)d_in[16];
    const float* csw  = (const float*)d_in[17]; const float* csb  = (const float*)d_in[18];
    const float* sdf  = (const float*)d_in[19];
    const float* stp  = (const float*)d_in[20];
    float* out = (float*)d_out;

    char* ws = (char*)d_ws;
    auto carve = [&](size_t bytes) -> void* {
        void* p = ws;
        ws += (bytes + 255) & ~(size_t)255;
        return p;
    };
    constexpr size_t HWs = 96 * 96;
    _Float16* ef16 = (_Float16*)carve((size_t)4 * 192 * HWs * 2);
    _Float16* buf0 = (_Float16*)carve((size_t)4 * 128 * HWs * 2);
    _Float16* buf1 = (_Float16*)carve((size_t)4 * 128 * HWs * 2);
    _Float16* fps  = (_Float16*)carve((size_t)4 * 128 * HW * 2);
    float*    feat = (float*)carve((size_t)4 * 38 * HW * 4);
    float*    smp  = (float*)carve((size_t)8 * 9 * HW * 4);

    // feature extraction trunk (WMMA implicit-GEMM)
    {
        int n4 = (4 * 192 * (int)HWs) / 4;
        cvt_f16_kernel<<<(n4 + 255) / 256, 256, 0, stream>>>(ef, ef16, n4);
    }
    conv3x3_wmma<192, 128, 96, 96, 0><<<dim3(4, 144, 4), 256, 0, stream>>>(ef16, cf0w, cf0b, buf0, nullptr);
    conv3x3_wmma<128, 128, 96, 96, 0><<<dim3(4, 144, 4), 256, 0, stream>>>(buf0, cf1w, cf1b, buf1, nullptr);
    conv3x3_wmma<128, 128, 96, 96, 0><<<dim3(4, 144, 4), 256, 0, stream>>>(buf1, cf2w, cf2b, buf0, nullptr);
    conv3x3_wmma<128, 512, 96, 96, 1><<<dim3(16, 144, 4), 256, 0, stream>>>(buf0, psw, psb, fps, nullptr);
    conv3x3_wmma<128, 38, 192, 192, 2><<<dim3(2, 576, 4), 256, 0, stream>>>(fps, cf3w, cf3b, nullptr, feat);

    // deformable sampling + heads
    deform_kernel<<<(8 * 9 * (int)HW + 255) / 256, 256, 0, stream>>>(x, feat, smp);
    weights_kernel<<<(4 * (int)HW + 255) / 256, 256, 0, stream>>>(feat, out);
    cur_kernel<<<(16 * (int)HW + 255) / 256, 256, 0, stream>>>(xcur, ccw, ccb, sdf, out);
    xall_kernel<<<(16 * (int)HW + 255) / 256, 256, 0, stream>>>(xcur, smp, ctw, ctb, stp, out);
    surround_kernel<<<(8 * (int)HW + 255) / 256, 256, 0, stream>>>(smp, csw, csb, sdf, out);
}
